// DMFMLayer_32212254720200
// MI455X (gfx1250) — matrix operand order repaired and checked by reading, not compiled, not run on hardware
//
#include <hip/hip_runtime.h>
#include <stdint.h>

// ---------------- types ----------------
typedef __attribute__((ext_vector_type(16))) __bf16 v16bf;
typedef __attribute__((ext_vector_type(8)))  __bf16 v8bf;
typedef __attribute__((ext_vector_type(8)))  float  v8f;
typedef __attribute__((ext_vector_type(4)))  unsigned int u32x4;
typedef __attribute__((ext_vector_type(8)))  int    i32x8;
typedef __attribute__((ext_vector_type(4)))  int    i32x4;

#define LSEQ   4096      // L = W*H
#define NBATCH 4
#define NCH    128       // C
#define DI     256       // d_inner
#define DZ     512       // 2*d_inner
#define NP     16384     // B*L positions
#define M2     32768     // 2 branches * NP
#define XD     40        // dt_rank + 2*D_STATE
#define NSTATE 16
#define SCHUNK 64        // scan TDM chunk (rows of B/C coeffs per TDM tile)

// ---------------- fp32 -> bf16 convert ----------------
__global__ void cvt_bf16_kernel(const float* __restrict__ src, __bf16* __restrict__ dst, int n) {
    int i = blockIdx.x * 256 + threadIdx.x;
    if (i < n) dst[i] = (__bf16)src[i];
}

// ---------------- block mean/var over 128 lanes ----------------
__device__ __forceinline__ float2 blk_meanvar128(float v, float* sbuf) {
    const int t = threadIdx.x;
    sbuf[t] = v; __syncthreads();
    #pragma unroll
    for (int s = 64; s > 0; s >>= 1) { if (t < s) sbuf[t] += sbuf[t + s]; __syncthreads(); }
    float mu = sbuf[0] * (1.0f / 128.0f);
    __syncthreads();
    float dv = v - mu;
    sbuf[t] = dv * dv; __syncthreads();
    #pragma unroll
    for (int s = 64; s > 0; s >>= 1) { if (t < s) sbuf[t] += sbuf[t + s]; __syncthreads(); }
    float var = sbuf[0] * (1.0f / 128.0f);
    __syncthreads();
    return make_float2(mu, var);
}

// ---------------- LN of x (branch 0) and channel-shuffled x (branch 1) ----------------
// writes fp32 (for residual) and bf16 (WMMA A operand), rows: [br*NP + p] x 128
__global__ __launch_bounds__(128) void ln_shuffle_kernel(
    const float* __restrict__ x, const float* __restrict__ gamma, const float* __restrict__ beta,
    float* __restrict__ x12f, __bf16* __restrict__ x12h)
{
    __shared__ float sbuf[128];
    const int p = blockIdx.x;            // 0..NP-1
    const int c = threadIdx.x;           // channel
    const int b = p >> 12;               // p / 4096
    const int l = p & (LSEQ - 1);
    const float g = gamma[c], be = beta[c];

    float v0 = x[((size_t)(b * NCH + c)) * LSEQ + l];
    const int src = ((c & 7) << 4) | (c >> 3);   // shuffled source channel
    float v1 = x[((size_t)(b * NCH + src)) * LSEQ + l];

    float2 mv0 = blk_meanvar128(v0, sbuf);
    float r0 = (v0 - mv0.x) * rsqrtf(mv0.y + 1e-5f) * g + be;
    x12f[(size_t)p * NCH + c] = r0;
    x12h[(size_t)p * NCH + c] = (__bf16)r0;

    float2 mv1 = blk_meanvar128(v1, sbuf);
    float r1 = (v1 - mv1.x) * rsqrtf(mv1.y + 1e-5f) * g + be;
    x12f[(size_t)(NP + p) * NCH + c] = r1;
    x12h[(size_t)(NP + p) * NCH + c] = (__bf16)r1;
}

// ---------------- WMMA GEMM: C[M,N] = A[M,K] * Bw[N,K]^T, bf16 in / f32 out --------------
// block = 128 threads = 4 waves; each wave owns a 16x64 tile (A-fragment reuse x4).
__global__ __launch_bounds__(128) void wmma_gemm_bf16(
    const __bf16* __restrict__ A, const __bf16* __restrict__ Bw,
    float* __restrict__ C, int M, int N, int K)
{
    const int lane = threadIdx.x & 31;
    const int wave = threadIdx.x >> 5;
    const int half = lane >> 4;          // 0: lanes 0-15, 1: lanes 16-31
    const int r    = lane & 15;
    const int tm   = blockIdx.y * 64 + wave * 16;
    const int tn   = blockIdx.x * 64;

    const __bf16* arow = A + (size_t)(tm + r) * K;
    v8f acc0 = {}, acc1 = {}, acc2 = {}, acc3 = {};

    for (int k0 = 0; k0 < K; k0 += 32) {
        // A fragment (16x32 bf16): half 0 holds K {0..7,16..23}, half 1 holds {8..15,24..31}
        v8bf alo = *reinterpret_cast<const v8bf*>(arow + k0 + half * 8);
        v8bf ahi = *reinterpret_cast<const v8bf*>(arow + k0 + 16 + half * 8);
        v16bf a = __builtin_shufflevector(alo, ahi, 0,1,2,3,4,5,6,7,8,9,10,11,12,13,14,15);
        // B fragment (32x16 bf16): half 0 holds K {0..7,16..23}, half 1 holds {8..15,24..31} of column n
        #pragma unroll
        for (int j = 0; j < 4; ++j) {
            const __bf16* brow = Bw + (size_t)(tn + j * 16 + r) * K + k0 + half * 8;
            v8bf blo = *reinterpret_cast<const v8bf*>(brow);
            v8bf bhi = *reinterpret_cast<const v8bf*>(brow + 16);
            v16bf bfr = __builtin_shufflevector(blo, bhi, 0,1,2,3,4,5,6,7,8,9,10,11,12,13,14,15);
            v8f* accp = (j == 0) ? &acc0 : (j == 1) ? &acc1 : (j == 2) ? &acc2 : &acc3;
            *accp = __builtin_amdgcn_wmma_f32_16x16x32_bf16(
                false, a, false, bfr, (short)0, *accp, false, false);
        }
    }
    // C/D layout: VGPR i -> row tm + half*8 + i, col tn + j*16 + r
    float* crow = C + (size_t)(tm + half * 8) * N + tn + r;
    #pragma unroll
    for (int i = 0; i < 8; ++i) {
        crow[(size_t)i * N + 0]  = acc0[i];
        crow[(size_t)i * N + 16] = acc1[i];
        crow[(size_t)i * N + 32] = acc2[i];
        crow[(size_t)i * N + 48] = acc3[i];
    }
}

// ---------------- causal depthwise conv (D_CONV=4) + SiLU ----------------
__global__ void conv_silu_kernel(const float* __restrict__ xz, const float* __restrict__ cw,
                                 const float* __restrict__ cb, float* __restrict__ u)
{
    int gid  = blockIdx.x * 256 + threadIdx.x;   // over M2*DI
    int d    = gid & (DI - 1);
    int rest = gid >> 8;                          // row 0..M2-1
    int l    = rest & (LSEQ - 1);
    int bb   = rest >> 12;                        // branch*4 + batch
    float acc = cb[d];
    #pragma unroll
    for (int j = 0; j < 4; ++j) {
        int ll = l - 3 + j;
        if (ll >= 0) acc += cw[d * 4 + j] * xz[((size_t)(bb * LSEQ + ll)) * DZ + d];
    }
    float sg = 1.0f / (1.0f + __expf(-acc));
    u[(size_t)rest * DI + d] = acc * sg;
}

// ---------------- x_dbl = u @ W_x^T  (N=40, K=256) ----------------
__global__ __launch_bounds__(64) void xdbl_kernel(const float* __restrict__ u,
                                                  const float* __restrict__ Wx,
                                                  float* __restrict__ xdbl)
{
    __shared__ float su[DI];
    const int m = blockIdx.x;
    for (int k = threadIdx.x; k < DI; k += 64) su[k] = u[(size_t)m * DI + k];
    __syncthreads();
    const int n = threadIdx.x;
    if (n < XD) {
        float acc = 0.0f;
        const float* w = Wx + n * DI;
        #pragma unroll 8
        for (int k = 0; k < DI; ++k) acc += su[k] * w[k];
        xdbl[(size_t)m * XD + n] = acc;
    }
}

// ---------------- dt = softplus(x_dbl[:, :8] @ W_dt^T + b_dt) ----------------
__global__ void dt_kernel(const float* __restrict__ xdbl, const float* __restrict__ Wdt,
                          const float* __restrict__ bdt, float* __restrict__ dt)
{
    int gid = blockIdx.x * 256 + threadIdx.x;   // over M2*DI
    int d = gid & (DI - 1);
    int m = gid >> 8;
    float acc = bdt[d];
    #pragma unroll
    for (int r = 0; r < 8; ++r) acc += xdbl[(size_t)m * XD + r] * Wdt[d * 8 + r];
    dt[(size_t)m * DI + d] = (acc > 20.0f) ? acc : __logf(1.0f + __expf(acc));
}

// ---------------- TDM issue: strided 2D tile (32 floats wide, SCHUNK rows) -> LDS ------
__device__ __forceinline__ void tdm_issue_bc(const float* gptr, unsigned int lds_off)
{
    uint64_t ga = (uint64_t)(uintptr_t)gptr;
    u32x4 g0 = { 1u,                                   // count=1, user mode
                 lds_off,                              // lds_addr (bytes)
                 (unsigned int)(ga & 0xFFFFFFFFu),
                 (unsigned int)((ga >> 32) & 0x1FFFFFFu) | (2u << 30) };  // type=2 "image"
    const unsigned int tdim0 = 1u << 20, tdim1 = 1u << 20;  // large: no OOB clipping
    const unsigned int tile0 = 32, tile1 = SCHUNK;
    const uint64_t     str0  = XD;                          // row stride in elements
    i32x8 g1 = { (int)(2u << 16),                              // data_size = 4B
                 (int)((tdim0 & 0xFFFFu) << 16),               // tensor_dim0 lo16
                 (int)((tdim0 >> 16) | ((tdim1 & 0xFFFFu) << 16)),
                 (int)((tdim1 >> 16) | (tile0 << 16)),         // tensor_dim1 hi16 | tile_dim0
                 (int)tile1,                                   // tile_dim1 (tile_dim2=0)
                 (int)(unsigned int)(str0 & 0xFFFFFFFFu),      // tensor_dim0_stride lo32
                 (int)(unsigned int)((str0 >> 32) & 0xFFFFu),  // stride0 hi16 | stride1 lo16
                 0 };
    i32x4 gz4 = { 0, 0, 0, 0 };
    i32x8 gz8 = { 0, 0, 0, 0, 0, 0, 0, 0 };
    // clang-23 / therock-10.0 arity: (g0, g1, g2, g3, g4, cpol)
    __builtin_amdgcn_tensor_load_to_lds(g0, g1, gz4, gz4, gz8, 0);
}

// ---------------- selective scan, TDM-fed, gating fused ----------------
// grid = 8 blocks (branch*4 + batch), block = 256 threads (one per d).
__global__ __launch_bounds__(256) void scan_kernel(
    const float* __restrict__ dtp, const float* __restrict__ up,
    const float* __restrict__ xdbl, const float* __restrict__ xz,
    const float* __restrict__ A_log, const float* __restrict__ D_skip,
    __bf16* __restrict__ yh)
{
    __shared__ float ldsBC[2][SCHUNK * 32];   // per chunk: 64 rows x (16 B | 16 C)
    const int bb = blockIdx.x;                // 0..7
    const int d  = threadIdx.x;               // 0..255
    const int r0 = bb * LSEQ;                 // base row in M2 space

    float A_d[NSTATE], h[NSTATE];
    #pragma unroll
    for (int n = 0; n < NSTATE; ++n) { A_d[n] = -__expf(A_log[d * NSTATE + n]); h[n] = 0.0f; }
    const float Dsk = D_skip[d];

    const int nch = LSEQ / SCHUNK;
    if (threadIdx.x < 32)   // wave 0 drives the TDM
        tdm_issue_bc(xdbl + (size_t)r0 * XD + 8, (unsigned int)(uintptr_t)&ldsBC[0][0]);

    for (int c = 0; c < nch; ++c) {
        if (threadIdx.x < 32) {
            if (c + 1 < nch) {
                tdm_issue_bc(xdbl + (size_t)(r0 + (c + 1) * SCHUNK) * XD + 8,
                             (unsigned int)(uintptr_t)&ldsBC[(c + 1) & 1][0]);
                __builtin_amdgcn_s_wait_tensorcnt(1);   // chunk c ready, c+1 in flight
            } else {
                __builtin_amdgcn_s_wait_tensorcnt(0);
            }
        }
        __syncthreads();
        const float* bc = &ldsBC[c & 1][0];
        int l = c * SCHUNK;
        for (int li = 0; li < SCHUNK; ++li, ++l) {
            const size_t row = (size_t)(r0 + l);
            float dtv = dtp[row * DI + d];
            float uv  = up [row * DI + d];
            float xbu = dtv * uv;
            float y = 0.0f;
            const float* bcl = bc + li * 32;
            #pragma unroll
            for (int n = 0; n < NSTATE; ++n) {
                float dA = __expf(dtv * A_d[n]);
                h[n] = dA * h[n] + xbu * bcl[n];
                y += h[n] * bcl[16 + n];
            }
            float zv = xz[row * DZ + DI + d];
            float sg = 1.0f / (1.0f + __expf(-zv));
            yh[row * DI + d] = (__bf16)((y + uv * Dsk) * (zv * sg));
        }
        __syncthreads();
    }
}

// ---------------- combine branches + residuals, LN, emit bf16 ----------------
__global__ __launch_bounds__(128) void combine_ln_kernel(
    const float* __restrict__ gy, const float* __restrict__ x12f,
    const float* __restrict__ s1p, const float* __restrict__ s2p,
    const float* __restrict__ gamma, const float* __restrict__ beta,
    __bf16* __restrict__ xmh)
{
    __shared__ float sbuf[128];
    const int p = blockIdx.x, c = threadIdx.x;
    const size_t i0 = (size_t)p * NCH + c;
    const size_t i1 = (size_t)(NP + p) * NCH + c;
    float v = gy[i0] + gy[i1] + x12f[i0] * s1p[0] + x12f[i1] * s2p[0];
    float2 mv = blk_meanvar128(v, sbuf);
    float r = (v - mv.x) * rsqrtf(mv.y + 1e-5f) * gamma[c] + beta[c];
    xmh[i0] = (__bf16)r;
}

// ---------------- transpose (B,L,C)->(B,C,L) with bias ----------------
__global__ void epilogue_kernel(const float* __restrict__ t, const float* __restrict__ bp,
                                float* __restrict__ out)
{
    int gid  = blockIdx.x * 256 + threadIdx.x;   // over NP*128
    int l    = gid & (LSEQ - 1);
    int rest = gid >> 12;
    int n    = rest & 127;
    int b    = rest >> 7;
    out[((size_t)(b * 128 + n)) * LSEQ + l] = t[((size_t)(b * LSEQ + l)) * NCH + n] + bp[n];
}

// ---------------- launch ----------------
extern "C" void kernel_launch(void* const* d_in, const int* in_sizes, int n_in,
                              void* d_out, int out_size, void* d_ws, size_t ws_size,
                              hipStream_t stream)
{
    (void)in_sizes; (void)n_in; (void)out_size; (void)ws_size;
    const float* x      = (const float*)d_in[0];
    const float* gamma  = (const float*)d_in[1];
    const float* beta   = (const float*)d_in[2];
    const float* W_in   = (const float*)d_in[3];
    const float* conv_w = (const float*)d_in[4];
    const float* conv_b = (const float*)d_in[5];
    const float* W_x    = (const float*)d_in[6];
    const float* W_dt   = (const float*)d_in[7];
    const float* b_dt   = (const float*)d_in[8];
    const float* A_log  = (const float*)d_in[9];
    const float* D_skip = (const float*)d_in[10];
    const float* W_out  = (const float*)d_in[11];
    const float* W_p    = (const float*)d_in[12];
    const float* b_p    = (const float*)d_in[13];
    const float* s1     = (const float*)d_in[14];
    const float* s2     = (const float*)d_in[15];
    float* out = (float*)d_out;

    char* ws = (char*)d_ws;
    size_t off = 0;
    auto take = [&](size_t bytes) -> char* {
        char* r = ws + off;
        off = (off + bytes + 255) & ~(size_t)255;
        return r;
    };
    float*  x12f = (float*) take((size_t)M2 * NCH * 4);
    float*  xz   = (float*) take((size_t)M2 * DZ  * 4);
    float*  u    = (float*) take((size_t)M2 * DI  * 4);
    float*  xdbl = (float*) take((size_t)M2 * XD  * 4);
    float*  dtb  = (float*) take((size_t)M2 * DI  * 4);
    __bf16* x12h = (__bf16*)take((size_t)M2 * NCH * 2);
    __bf16* yh   = (__bf16*)take((size_t)M2 * DI  * 2);
    __bf16* xmh  = (__bf16*)take((size_t)NP * NCH * 2);
    __bf16* Wih  = (__bf16*)take((size_t)DZ  * NCH * 2);
    __bf16* Woh  = (__bf16*)take((size_t)NCH * DI  * 2);
    __bf16* Wph  = (__bf16*)take((size_t)NCH * NCH * 2);
    float* gy   = xz;   // xz dead after scan -> reuse for W_out GEMM result
    float* tmpo = u;    // u dead after scan  -> reuse for final GEMM result

    cvt_bf16_kernel<<<(DZ * NCH + 255) / 256, 256, 0, stream>>>(W_in,  Wih, DZ * NCH);
    cvt_bf16_kernel<<<(NCH * DI + 255) / 256, 256, 0, stream>>>(W_out, Woh, NCH * DI);
    cvt_bf16_kernel<<<(NCH * NCH + 255) / 256, 256, 0, stream>>>(W_p,  Wph, NCH * NCH);

    ln_shuffle_kernel<<<NP, 128, 0, stream>>>(x, gamma, beta, x12f, x12h);

    wmma_gemm_bf16<<<dim3(DZ / 64, M2 / 64), 128, 0, stream>>>(x12h, Wih, xz, M2, DZ, NCH);

    conv_silu_kernel<<<(M2 * DI) / 256, 256, 0, stream>>>(xz, conv_w, conv_b, u);
    xdbl_kernel<<<M2, 64, 0, stream>>>(u, W_x, xdbl);
    dt_kernel<<<(M2 * DI) / 256, 256, 0, stream>>>(xdbl, W_dt, b_dt, dtb);

    scan_kernel<<<8, 256, 0, stream>>>(dtb, u, xdbl, xz, A_log, D_skip, yh);

    wmma_gemm_bf16<<<dim3(NCH / 64, M2 / 64), 128, 0, stream>>>(yh, Woh, gy, M2, NCH, DI);

    combine_ln_kernel<<<NP, 128, 0, stream>>>(gy, x12f, s1, s2, gamma, beta, xmh);

    wmma_gemm_bf16<<<dim3(NCH / 64, NP / 64), 128, 0, stream>>>(xmh, Wph, tmpo, NP, NCH, NCH);

    epilogue_kernel<<<(NP * 128) / 256, 256, 0, stream>>>(tmpo, b_p, out);
}
